// model_51367808860191
// MI455X (gfx1250) — matrix operand order repaired and checked by reading, not compiled, not run on hardware
//
#include <hip/hip_runtime.h>
#include <hip/hip_bf16.h>

// ---------------------------------------------------------------------------
// Fused per-batch-element pipeline for the band-gated SPD graph model.
// MI455X (gfx1250) design notes:
//  * memory-bound: 157MB of pcc/A inputs dominate; everything else fused in LDS
//  * one 256-thread workgroup (8 wave32) per batch element, ~300KB dynamic LDS
//  * phase-A fills use CDNA5 async global->LDS DMA (ASYNCcnt-tracked):
//      - A[b]   : contiguous  global_load_async_to_lds_b128
//      - pcc[b] : band-transposing scatter via global_load_async_to_lds_b32
//  * all 62x62x62 GEMMs (covariance xc^T xc, branch com@lw^T) run on
//    v_wmma_f32_16x16x32_f16 with 64-padded tiles, K in two 32-steps;
//    operands staged in K-contiguous LDS layouts so each WMMA operand is
//    1-2 vector LDS loads per lane (no scalar b16 packing)
//  * trace-normalization folded: spd = (xc^T xc)/sum(xc^2) + eps*I
//  * branch affine folded into GEMM epilogue:
//      z = sigmoid(cw*(com@lw^T) + cb*rowsum(lw) + lb)
//  * BatchNorm needs a cross-B reduction -> separate stats + head kernels
// ---------------------------------------------------------------------------

typedef __attribute__((ext_vector_type(16))) _Float16 v16h;
typedef __attribute__((ext_vector_type(8)))  _Float16 v8h;
typedef __attribute__((ext_vector_type(8)))  float    v8f;

#define BDIM 256
#define BATCH 1024

constexpr int C    = 62;
constexpr int BAND = 5;
constexpr int CCsz = C * C;       // 3844
constexpr int CP   = 64;          // padded dim
constexpr int CP2  = CP * CP;     // 4096
constexpr int HID  = 10;
constexpr int NOUT = 16;

// ---- LDS arena byte offsets (aliasing is phase-disjoint) -------------------
constexpr int OFF_XF   = 0;                         // f32 [5][3844] gated fadj (phases A,B)
constexpr int OFF_XS   = OFF_XF  + 4 * BAND * CCsz; // f32 [5][3844] gated sadj
constexpr int OFF_Z    = 0;                         // alias: f16 z[2][5][4096] (phase C)
constexpr int OFF_COM  = OFF_XS  + 4 * BAND * CCsz; // f32 [5][4096] (row i, K-contig cols)
constexpr int OFF_XC   = OFF_COM + 4 * BAND * CP2;  // f16 [2][4096] xcT (col-major centered)
constexpr int OFF_HIDH = OFF_XC;                    // alias: f32 [62*64] hidden adjacency
constexpr int OFF_HIDE = OFF_XC  + 2 * 2 * CP2;     // f32 [62*64] essential adjacency
constexpr int OFF_LWT  = OFF_HIDE + 4 * C * CP;     // f16 [2][4096] lw row-major padded (B op)
constexpr int OFF_RED  = OFF_LWT + 2 * 2 * CP2;     // f32 [2560] reductions / xw1 / feat
constexpr int OFF_FEAT = OFF_RED + 4 * C * NOUT;    // alias inside RED region
constexpr int OFF_WSC  = OFF_RED + 4 * 10 * BDIM;   // f32 [2][5][64] attention scores
constexpr int OFF_AW1  = OFF_WSC + 4 * 2 * BAND * CP;
constexpr int OFF_ROWS = OFF_AW1 + 4 * HID * C;     // f32 [2][64] rowsum(lw)
constexpr int OFF_SGET = OFF_ROWS + 4 * 2 * CP;     // f32 [10] gate logits
constexpr int OFF_SSQ  = OFF_SGET + 4 * 10;         // f32 [2]
constexpr int OFF_AB1  = OFF_SSQ + 4 * 2;           // f32 [10]
constexpr int OFF_AW2  = OFF_AB1 + 4 * HID;         // f32 [10]
constexpr int SMEM_BYTES = OFF_AW2 + 4 * HID;       // ~300,240 B  (<= 320KB/WG)

__device__ __forceinline__ float sigmoidf_(float x) { return 1.f / (1.f + __expf(-x)); }

__global__ __launch_bounds__(BDIM) void fused_band_spd_kernel(
    const float* __restrict__ de,     const float* __restrict__ pcc,
    const float* __restrict__ Aadj,   const float* __restrict__ conv_w,
    const float* __restrict__ conv_b,
    const float* __restrict__ hconv_w, const float* __restrict__ hconv_b,
    const float* __restrict__ hlin_w,  const float* __restrict__ hlin_b,
    const float* __restrict__ econv_w, const float* __restrict__ econv_b,
    const float* __restrict__ elin_w,  const float* __restrict__ elin_b,
    const float* __restrict__ att_w1,  const float* __restrict__ att_b1,
    const float* __restrict__ att_w2,
    const float* __restrict__ gc_w,    const float* __restrict__ gc_b,
    const float* __restrict__ wpar,
    const float* __restrict__ fcw,     const float* __restrict__ fcb,
    float* __restrict__ zout)          // [BATCH][64] pre-BN logits
{
    extern __shared__ char smem[];
    float*    xf    = (float*)(smem + OFF_XF);
    float*    xs    = (float*)(smem + OFF_XS);
    _Float16* zbuf  = (_Float16*)(smem + OFF_Z);
    float*    com   = (float*)(smem + OFF_COM);
    _Float16* xc    = (_Float16*)(smem + OFF_XC);   // xcT: [which][c*CP + h]
    float*    hidh  = (float*)(smem + OFF_HIDH);
    float*    hide  = (float*)(smem + OFF_HIDE);
    _Float16* lwp   = (_Float16*)(smem + OFF_LWT);  // [v][o*CP + c]
    float*    red   = (float*)(smem + OFF_RED);
    float*    xw1   = (float*)(smem + OFF_RED);     // alias (phase D)
    float*    feat  = (float*)(smem + OFF_FEAT);    // alias (phase D)
    float*    wsc   = (float*)(smem + OFF_WSC);
    float*    aw1   = (float*)(smem + OFF_AW1);
    float*    rowsum= (float*)(smem + OFF_ROWS);
    float*    sget  = (float*)(smem + OFF_SGET);
    float*    ssq   = (float*)(smem + OFF_SSQ);
    float*    ab1   = (float*)(smem + OFF_AB1);
    float*    aw2v  = (float*)(smem + OFF_AW2);

    const int b    = blockIdx.x;
    const int t    = threadIdx.x;
    const int wave = t >> 5;
    const int lane = t & 31;
    const int lm   = lane & 15;   // M/N position within tile
    const int hf   = lane >> 4;   // lane half (K half select)

    // ===================== Phase A: async loads + gates =====================
    // A[b] is already band-major: contiguous async DMA, 16B per lane per issue
    {
        const char* gA = (const char*)(Aadj + (size_t)b * (BAND * CCsz));
        for (int m = t * 16; m < BAND * CCsz * 4; m += BDIM * 16) {
            unsigned ldsoff = (unsigned)(OFF_XS + m);
            const char* g = gA + m;
            asm volatile("global_load_async_to_lds_b128 %0, %1, off"
                         :: "v"(ldsoff), "v"(g) : "memory");
        }
    }
    // pcc[b] is [62,62,5]; band-transposing scatter DMA (per-lane LDS dest)
    {
        const char* gP = (const char*)(pcc + (size_t)b * (CCsz * BAND));
        for (int m = t; m < BAND * CCsz; m += BDIM) {
            int n = m / BAND, i = m % BAND;           // n = h*62+w, i = band
            unsigned ldsoff = (unsigned)(OFF_XF + (i * CCsz + n) * 4);
            const char* g = gP + (size_t)(n * BAND + i) * 4;
            asm volatile("global_load_async_to_lds_b32 %0, %1, off"
                         :: "v"(ldsoff), "v"(g) : "memory");
        }
    }

    // stage small weights while the DMA runs; zero-pad lw then fill
    for (int m = t; m < 2 * CP2; m += BDIM) lwp[m] = (_Float16)0.f;
    for (int m = t; m < HID * C; m += BDIM) aw1[m] = att_w1[m];
    if (t < HID) { ab1[t] = att_b1[t]; aw2v[t] = att_w2[t]; }
    if (t < 2 * CP) {   // rowsum(lw) per output channel (pads -> 0)
        int v = t / CP, o = t % CP;
        float s = 0.f;
        if (o < C) {
            const float* lw = v ? elin_w : hlin_w;
            for (int c = 0; c < C; ++c) s += lw[o * C + c];
        }
        rowsum[v * CP + o] = s;
    }
    __syncthreads();
    for (int m = t; m < C * C; m += BDIM) {
        int o = m / C, c = m % C;
        lwp[o * CP + c]       = (_Float16)hlin_w[m];   // B operand: K(c)-contiguous
        lwp[CP2 + o * CP + c] = (_Float16)elin_w[m];
    }
    // drain the async DMA before anyone touches xf/xs
    asm volatile("s_wait_asynccnt 0x0" ::: "memory");
    __syncthreads();

    // gate logits: s[o] = sum_{i,hw} x[i,hw] * conv_w[o,i,hw]
    float af[BAND] = {}, as_[BAND] = {};
    for (int m = t; m < BAND * CCsz; m += BDIM) {
        float vf = xf[m], vs = xs[m];
        __builtin_prefetch(conv_w + m + BDIM, 0, 3);   // stream conv_w (L2-resident)
        #pragma unroll
        for (int o = 0; o < BAND; ++o) {
            float w = conv_w[o * (BAND * CCsz) + m];
            af[o]  = fmaf(vf, w, af[o]);
            as_[o] = fmaf(vs, w, as_[o]);
        }
    }
    #pragma unroll
    for (int o = 0; o < BAND; ++o) {
        red[o * BDIM + t]          = af[o];
        red[(o + BAND) * BDIM + t] = as_[o];
    }
    __syncthreads();
    if (t < 2 * BAND) {
        float s = 0.f;
        for (int j = 0; j < BDIM; ++j) s += red[t * BDIM + j];
        sget[t] = s + conv_b[t % BAND];
    }
    __syncthreads();
    float gf[BAND], gs[BAND];
    {   // redundant per-thread softmax over the 5 bands (cheap)
        float mf = -1e30f, ms = -1e30f;
        for (int o = 0; o < BAND; ++o) { mf = fmaxf(mf, sget[o]); ms = fmaxf(ms, sget[o + BAND]); }
        float ef = 0.f, es = 0.f;
        for (int o = 0; o < BAND; ++o) {
            gf[o] = __expf(sget[o] - mf);        ef += gf[o];
            gs[o] = __expf(sget[o + BAND] - ms); es += gs[o];
        }
        for (int o = 0; o < BAND; ++o) { gf[o] /= ef; gs[o] /= es; }
    }
    for (int m = t; m < BAND * CCsz; m += BDIM) {   // relu(g*x) in place
        int i = m / CCsz;
        xf[m] = fmaxf(gf[i] * xf[m], 0.f);
        xs[m] = fmaxf(gs[i] * xs[m], 0.f);
    }
    __syncthreads();

    // ========== Phase B: per-band SPD  com = 0.5*(cf/ssqf + cs/ssqs)+eps*I ==
    for (int k = 0; k < BAND; ++k) {
        for (int m = t; m < 2 * CP2; m += BDIM) xc[m] = (_Float16)0.f;
        __syncthreads();
        float ssqp = 0.f;
        if (t < 2 * C) {        // row-center + f16 pack (transposed) + sum-sq
            int which = t / C, r = t % C;
            const float* src = (which ? xs : xf) + k * CCsz + r * C;
            _Float16* dstT = xc + which * CP2 + r;   // column r of xcT
            float mu = 0.f;
            for (int w = 0; w < C; ++w) mu += src[w];
            mu *= (1.f / C);
            float sq = 0.f;
            for (int w = 0; w < C; ++w) {
                float v = src[w] - mu;
                dstT[w * CP] = (_Float16)v;          // xcT[w][r] = xc[r][w]... see note
                sq = fmaf(v, v, sq);
            }
            ssqp = sq;
        }
        red[t] = ssqp;
        __syncthreads();
        if (t < 2) {
            float s = 0.f;
            for (int r = 0; r < C; ++r) s += red[t * C + r];
            ssq[t] = s;
        }
        __syncthreads();
        // Note: cov[j,l] = sum_h xc[h,j]*xc[h,l]; we stored xcT[c][h]=xc[h][c],
        // so both A (row j) and B (row l) operands are K(h)-contiguous rows.
        float inv_f = 1.f / ssq[0];
        float inv_s = 1.f / ssq[1];

        for (int tile = wave; tile < 16; tile += 8) {
            int tj = tile >> 2, tl = tile & 3;
            const _Float16* aFp = xc + (tj * 16 + lm) * CP;
            const _Float16* bFp = xc + (tl * 16 + lm) * CP;
            const _Float16* aSp = aFp + CP2;
            const _Float16* bSp = bFp + CP2;
            v8f accf = {}, accs = {};
            #pragma unroll
            for (int kk = 0; kk < 2; ++kk) {
                int k0 = kk * 32;
                // A layout: K = k0 + 16*hf + e
                v8h alo = *(const v8h*)(aFp + k0 + 16 * hf);
                v8h ahi = *(const v8h*)(aFp + k0 + 16 * hf + 8);
                v16h a_f = __builtin_shufflevector(alo, ahi,
                           0,1,2,3,4,5,6,7,8,9,10,11,12,13,14,15);
                v8h slo = *(const v8h*)(aSp + k0 + 16 * hf);
                v8h shi = *(const v8h*)(aSp + k0 + 16 * hf + 8);
                v16h a_s = __builtin_shufflevector(slo, shi,
                           0,1,2,3,4,5,6,7,8,9,10,11,12,13,14,15);
                // B layout: K = k0 + 16*hf + e  -> one 32B contiguous load
                v16h b_f = *(const v16h*)(bFp + k0 + 16 * hf);
                v16h b_s = *(const v16h*)(bSp + k0 + 16 * hf);
                accf = __builtin_amdgcn_wmma_f32_16x16x32_f16(false, a_f, false, b_f,
                                                              (short)0, accf, false, false);
                accs = __builtin_amdgcn_wmma_f32_16x16x32_f16(false, a_s, false, b_s,
                                                              (short)0, accs, false, false);
            }
            #pragma unroll
            for (int p = 0; p < 8; ++p) {       // C/D layout: M = p + 8*half
                int row = tj * 16 + p + 8 * hf;
                int col = tl * 16 + lm;
                float v = 0.5f * (accf[p] * inv_f + accs[p] * inv_s);
                if (row == col) v += 1e-5f;
                com[k * CP2 + row * CP + col] = v;
            }
        }
        __syncthreads();
    }

    // ====== Phase C: branches z = sigmoid(cw*(com@lw^T)+cb*rowsum+lb) =======
    for (int v = 0; v < 2; ++v) {
        const float* cw = v ? econv_w : hconv_w;
        const float* cb = v ? econv_b : hconv_b;
        const float* lb = v ? elin_b : hlin_b;
        const _Float16* lwv = lwp + v * CP2;
        _Float16* zv = zbuf + v * (BAND * CP2);
        for (int k = 0; k < BAND; ++k) {
            float cwk = cw[k], cbk = cb[k];
            for (int tile = wave; tile < 16; tile += 8) {
                int tj = tile >> 2, tl = tile & 3;
                const float*    pA = com + k * CP2 + (tj * 16 + lm) * CP;  // K-contig
                const _Float16* pB = lwv + (tl * 16 + lm) * CP;            // K-contig
                v8f acc = {};
                #pragma unroll
                for (int kk = 0; kk < 2; ++kk) {
                    int k0 = kk * 32;
                    v8f flo = *(const v8f*)(pA + k0 + 16 * hf);
                    v8f fhi = *(const v8f*)(pA + k0 + 16 * hf + 8);
                    v16h a;
                    #pragma unroll
                    for (int e = 0; e < 8; ++e) {
                        a[e]     = (_Float16)flo[e];
                        a[e + 8] = (_Float16)fhi[e];
                    }
                    v16h bm = *(const v16h*)(pB + k0 + 16 * hf);
                    acc = __builtin_amdgcn_wmma_f32_16x16x32_f16(false, a, false, bm,
                                                                 (short)0, acc, false, false);
                }
                int col = tl * 16 + lm;
                float bias = (col < C) ? fmaf(cbk, rowsum[v * CP + col], lb[col]) : 0.f;
                #pragma unroll
                for (int p = 0; p < 8; ++p) {
                    int row = tj * 16 + p + 8 * hf;
                    zv[k * CP2 + row * CP + col] =
                        (_Float16)sigmoidf_(fmaf(cwk, acc[p], bias));
                }
            }
        }
    }
    __syncthreads();

    // attention scores w[v,k,i] = sum_h tanh(z[k,i,:]@w1[h,:]+b1[h]) * w2[h]
    for (int task = t; task < 2 * BAND * C; task += BDIM) {
        int v = task / (BAND * C);
        int rem = task % (BAND * C);
        int k = rem / C, i = rem % C;
        const _Float16* zrow = zbuf + v * (BAND * CP2) + k * CP2 + i * CP;
        float sc = 0.f;
        for (int hh = 0; hh < HID; ++hh) {
            float s1 = ab1[hh];
            for (int c = 0; c < C; ++c) s1 = fmaf((float)zrow[c], aw1[hh * C + c], s1);
            sc = fmaf(tanhf(s1), aw2v[hh], sc);
        }
        wsc[v * BAND * CP + k * CP + i] = sc;
    }
    __syncthreads();
    // beta = softmax_k(score); adjacency L[i,c] = sum_k beta[k,i]*z[k,i,c]
    for (int task = t; task < 2 * C; task += BDIM) {
        int v = task / C, i = task % C;
        float s[BAND], m = -1e30f;
        for (int k = 0; k < BAND; ++k) { s[k] = wsc[v * BAND * CP + k * CP + i]; m = fmaxf(m, s[k]); }
        float tot = 0.f;
        for (int k = 0; k < BAND; ++k) { s[k] = __expf(s[k] - m); tot += s[k]; }
        float inv = 1.f / tot;
        float* hv = v ? hide : hidh;
        const _Float16* zv = zbuf + v * (BAND * CP2);
        for (int c = 0; c < C; ++c) {
            float acc = 0.f;
            for (int k = 0; k < BAND; ++k)
                acc = fmaf(s[k], (float)zv[k * CP2 + i * CP + c], acc);
            hv[i * CP + c] = acc * inv;
        }
    }
    __syncthreads();

    // ============ Phase D: Chebyshev GCN + feature FC -> z logits ===========
    const float* deb = de + (size_t)b * (C * BAND);
    for (int m = t; m < C * NOUT; m += BDIM) {      // xw1 = de @ gc_w[1]
        int j = m / NOUT, f = m % NOUT;
        float s = 0.f;
        #pragma unroll
        for (int q = 0; q < BAND; ++q)
            s = fmaf(deb[j * BAND + q], gc_w[(BAND + q) * NOUT + f], s);
        xw1[m] = s;
    }
    __syncthreads();
    {
        float e0 = __expf(wpar[0]);
        float wnorm = e0 / e0;                      // single-element softmax == 1
        for (int m = t; m < C * NOUT; m += BDIM) {
            int i = m / NOUT, f = m % NOUT;
            float x0 = 0.f;
            #pragma unroll
            for (int q = 0; q < BAND; ++q)
                x0 = fmaf(deb[i * BAND + q], gc_w[q * NOUT + f], x0);
            float base = x0 + gc_b[f] + gc_b[NOUT + f];
            float ah = 0.f, ae = 0.f;
            for (int j = 0; j < C; ++j) {
                float w1v = xw1[j * NOUT + f];
                ah = fmaf(hidh[i * CP + j], w1v, ah);
                ae = fmaf(hide[i * CP + j], w1v, ae);
            }
            float oh = fmaxf(base + ah, 0.f);
            float oe = fmaxf(base + ae, 0.f);
            feat[m] = fmaxf(0.5f * (oh + oe) * wnorm, 0.f);
        }
    }
    __syncthreads();
    if (t < C) {                                    // z[o] = feat @ fc_w[o,:] + b
        float s = fcb[t];
        const float* w = fcw + t * (C * NOUT);
        for (int n = 0; n < C * NOUT; ++n) s = fmaf(feat[n], w[n], s);
        zout[b * CP + t] = s;
    }
}

// ================= Phase 2: BatchNorm batch statistics ======================
__global__ __launch_bounds__(256) void bn_stats_kernel(const float* __restrict__ zbuf,
                                                       float* __restrict__ stats)
{
    __shared__ float s1[256], s2[256];
    const int o = blockIdx.x, t = threadIdx.x;
    float a = 0.f, b2 = 0.f;
    for (int bb = t; bb < BATCH; bb += 256) {
        float v = zbuf[bb * CP + o];
        a += v;
        b2 = fmaf(v, v, b2);
    }
    s1[t] = a; s2[t] = b2;
    __syncthreads();
    for (int off = 128; off > 0; off >>= 1) {
        if (t < off) { s1[t] += s1[t + off]; s2[t] += s2[t + off]; }
        __syncthreads();
    }
    if (t == 0) {
        float mean = s1[0] * (1.f / BATCH);
        float var  = s2[0] * (1.f / BATCH) - mean * mean;   // biased (jnp.var)
        stats[o] = mean;
        stats[CP + o] = var;
    }
}

// ================= Phase 3: BN + sigmoid + final FC(62->2) ==================
__global__ __launch_bounds__(64) void head_kernel(const float* __restrict__ zbuf,
                                                  const float* __restrict__ stats,
                                                  const float* __restrict__ bn_g,
                                                  const float* __restrict__ bn_b,
                                                  const float* __restrict__ fc4_w,
                                                  const float* __restrict__ fc4_b,
                                                  float* __restrict__ out)
{
    __shared__ float o1[C];
    const int b = blockIdx.x, t = threadIdx.x;
    if (t < C) {
        float z = zbuf[b * CP + t];
        float y = (z - stats[t]) * rsqrtf(stats[CP + t] + 1e-5f) * bn_g[t] + bn_b[t];
        float s = 1.f / (1.f + __expf(-y));
        out[b * C + t] = s;      // output1
        o1[t] = s;
    }
    __syncthreads();
    if (t < 2) {
        float s = fc4_b[t];
        for (int o = 0; o < C; ++o) s = fmaf(o1[o], fc4_w[t * C + o], s);
        out[BATCH * C + b * 2 + t] = s;   // output
    }
}

extern "C" void kernel_launch(void* const* d_in, const int* in_sizes, int n_in,
                              void* d_out, int out_size, void* d_ws, size_t ws_size,
                              hipStream_t stream) {
    (void)in_sizes; (void)n_in; (void)out_size; (void)ws_size;
    const float* de      = (const float*)d_in[0];
    const float* pcc     = (const float*)d_in[1];
    const float* Aadj    = (const float*)d_in[2];
    const float* conv_w  = (const float*)d_in[3];
    const float* conv_b  = (const float*)d_in[4];
    const float* hconv_w = (const float*)d_in[5];
    const float* hconv_b = (const float*)d_in[6];
    const float* hlin_w  = (const float*)d_in[7];
    const float* hlin_b  = (const float*)d_in[8];
    const float* econv_w = (const float*)d_in[9];
    const float* econv_b = (const float*)d_in[10];
    const float* elin_w  = (const float*)d_in[11];
    const float* elin_b  = (const float*)d_in[12];
    const float* att_w1  = (const float*)d_in[13];
    const float* att_b1  = (const float*)d_in[14];
    const float* att_w2  = (const float*)d_in[15];
    const float* gc_w    = (const float*)d_in[16];
    const float* gc_b    = (const float*)d_in[17];
    const float* wpar    = (const float*)d_in[18];
    const float* fc_w    = (const float*)d_in[19];
    const float* fc_b    = (const float*)d_in[20];
    const float* bn_g    = (const float*)d_in[21];
    const float* bn_b    = (const float*)d_in[22];
    const float* fc4_w   = (const float*)d_in[23];
    const float* fc4_b   = (const float*)d_in[24];

    float* zbuf  = (float*)d_ws;          // [BATCH][64] pre-BN logits
    float* stats = zbuf + BATCH * CP;     // mean[64], var[64]

    fused_band_spd_kernel<<<BATCH, BDIM, SMEM_BYTES, stream>>>(
        de, pcc, Aadj, conv_w, conv_b,
        hconv_w, hconv_b, hlin_w, hlin_b,
        econv_w, econv_b, elin_w, elin_b,
        att_w1, att_b1, att_w2,
        gc_w, gc_b, wpar, fc_w, fc_b, zbuf);
    bn_stats_kernel<<<C, 256, 0, stream>>>(zbuf, stats);
    head_kernel<<<BATCH, 64, 0, stream>>>(zbuf, stats, bn_g, bn_b, fc4_w, fc4_b,
                                          (float*)d_out);
}
